// ExtractorMLP_7438883357621
// MI455X (gfx1250) — matrix operand order repaired and checked by reading, not compiled, not run on hardware
//
#include <hip/hip_runtime.h>

// ---------------------------------------------------------------------------
// Edge-attention MLP for MI455X (gfx1250), wave32 + v_wmma_f32_16x16x32_bf16.
//   out[e] = W3^T relu(W2^T relu(W1^T concat(emb[col[e]], emb[row[e]]) + b1) + b2) + b3
// Compute-bound (~315 GFLOP per launch); emb + all weights are L2-resident,
// so the design pushes everything through the bf16 WMMA pipe with f32 accum.
// ---------------------------------------------------------------------------

typedef __attribute__((ext_vector_type(8)))  float        v8f;
typedef __attribute__((ext_vector_type(16))) __bf16       v16bf;
typedef __attribute__((ext_vector_type(2)))  __bf16       v2bf;
typedef __attribute__((ext_vector_type(4)))  unsigned int v4u;

struct Frag32 { v4u lo, hi; };   // 32 bytes = one bf16 WMMA operand per lane

// Pack two f32 -> one dword of bf16 pair; backend emits v_cvt_pk_bf16_f32.
__device__ __forceinline__ unsigned int pack2bf(float x, float y) {
  v2bf t;
  t.x = (__bf16)x;
  t.y = (__bf16)y;
  return __builtin_bit_cast(unsigned int, t);
}

__device__ __forceinline__ unsigned short f2bf(float f) {
  __bf16 b = (__bf16)f;
  return __builtin_bit_cast(unsigned short, b);
}

__device__ __forceinline__ v16bf make_frag(v4u lo, v4u hi) {
  Frag32 f{lo, hi};
  return __builtin_bit_cast(v16bf, f);
}

// ---------------------------------------------------------------------------
// Pack a [K,N] f32 weight matrix into bf16 B-fragments for
// v_wmma_*_16x16x32_bf16.  Fragment (nt,kt) is 32 lanes x 16 bf16, stored
// contiguously so the GEMM kernel reads 32B/lane fully coalesced.
// B layout used here (K=32 x N=16): lane n%16 = column; lanes 0-15 hold K 0..15,
// lanes 16-31 hold K 16..31, K ascending through the lane's 16 values.
// ---------------------------------------------------------------------------
__global__ void pack_weights(const float* __restrict__ W,
                             unsigned short* __restrict__ outp,
                             int K, int N, int KT, int NT) {
  int idx = blockIdx.x * 256 + threadIdx.x;
  int total = NT * KT * 512;
  if (idx >= total) return;
  int j    = idx & 15;          // value index within lane
  int l    = (idx >> 4) & 31;   // lane
  int frag = idx >> 9;
  int kt   = frag % KT;
  int nt   = frag / KT;
  int k = kt * 32 + j + ((l < 16) ? 0 : 16);
  int n = nt * 16 + (l & 15);
  float v = (k < K && n < N) ? W[k * N + n] : 0.0f;   // N<16 (W3) zero-padded
  outp[idx] = f2bf(v);
}

// ---------------------------------------------------------------------------
// Main fused kernel: tile = 128 edges per workgroup, 8 waves, wave w owns
// edges [16w,16w+16) of the tile.  LDS A = concat features (bf16), per-wave
// slice Xbuf stages activations between layers (no inter-wave sync needed
// after the gather barrier; per-wave LDS ops are in-order on CDNA5).
// ---------------------------------------------------------------------------
#define HID 128
#define AP  264   // A row stride in bf16 (256+8): 132 dwords -> conflict-free
#define XP  136   // X slice row stride in bf16 (128+8): 68 dwords

__global__ __launch_bounds__(256)
void edge_mlp_wmma(const float* __restrict__ emb,
                   const int*   __restrict__ ei,
                   const unsigned short* __restrict__ W1p,
                   const unsigned short* __restrict__ W2p,
                   const unsigned short* __restrict__ W3p,
                   const float* __restrict__ b1,
                   const float* __restrict__ b2,
                   const float* __restrict__ b3,
                   float* __restrict__ out,
                   int E) {
  __shared__ __align__(16) unsigned short Abuf[128 * AP];      // 66 KB
  __shared__ __align__(16) unsigned short Xbuf[8][16 * XP];    // 34 KB

  const int tid = threadIdx.x;

  // ---- gather phase: one (edge,endpoint) row per thread, f32 -> bf16 ----
  {
    const int el = tid >> 1, half = tid & 1;
    const int eg = blockIdx.x * 128 + el;
    int node = 0;
    if (eg < E) node = half ? ei[E + eg] : ei[eg];
    const float4* src = (const float4*)(emb + (size_t)node * HID);
    unsigned short* dst = Abuf + el * AP + half * HID;
#pragma unroll
    for (int i = 0; i < 16; ++i) {
      float4 a = src[2 * i];
      float4 b = src[2 * i + 1];
      v4u p;
      p.x = pack2bf(a.x, a.y);
      p.y = pack2bf(a.z, a.w);
      p.z = pack2bf(b.x, b.y);
      p.w = pack2bf(b.z, b.w);
      *(v4u*)(dst + i * 8) = p;   // ds_store_b128
    }
  }
  __syncthreads();

  const int wave = tid >> 5;
  const int lane = tid & 31;
  const int lrow = lane & 15;   // A row / C-D column within tile
  const int lhi  = lane >> 4;

  unsigned short* xs = &Xbuf[wave][0];
  // A-fragment base: row (wave*16 + lrow); A layout used here: lanes<16 take
  // k in {0..15}, lanes>=16 take {16..31} -> two b128 chunks.
  const unsigned short* arow = Abuf + (wave * 16 + lrow) * AP + lhi * 16;

  v8f acc2[8];
#pragma unroll
  for (int i = 0; i < 8; ++i)
    acc2[i] = v8f{0.f, 0.f, 0.f, 0.f, 0.f, 0.f, 0.f, 0.f};

  // ---- fused layer1 (256->512, relu) + layer2 (512->128) accumulation ----
  for (int c2 = 0; c2 < 16; ++c2) {          // 32-column chunk of X1
#pragma unroll
    for (int ntl = 0; ntl < 2; ++ntl) {
      const int nt = c2 * 2 + ntl;           // layer-1 N tile (0..31)
      v8f acc1 = v8f{0.f, 0.f, 0.f, 0.f, 0.f, 0.f, 0.f, 0.f};
#pragma unroll
      for (int kb = 0; kb < 8; ++kb) {
        const unsigned short* ap = arow + kb * 32;
        v4u alo = *(const v4u*)(ap);
        v4u ahi = *(const v4u*)(ap + 8);
        const unsigned short* bp = W1p + (((nt * 8 + kb) * 32 + lane) << 4);
        v4u blo = *(const v4u*)(bp);
        v4u bhi = *(const v4u*)(bp + 8);
        acc1 = __builtin_amdgcn_wmma_f32_16x16x32_bf16(
            false, make_frag(alo, ahi), false, make_frag(blo, bhi),
            (short)0, acc1, false, false);
      }
      const float bias = b1[nt * 16 + lrow];
#pragma unroll
      for (int r = 0; r < 8; ++r) {          // C layout: col=lrow, row=r+8*lhi
        float v = fmaxf(acc1[r] + bias, 0.0f);
        xs[(r + lhi * 8) * XP + ntl * 16 + lrow] = f2bf(v);
      }
    }

    // layer-2: accumulate K-chunk c2 into persistent [16,128] accumulators.
    // Per-wave LDS ops are in-order; compiler inserts the precise dscnt wait
    // before the consuming WMMA.
    {
      const unsigned short* ap2 = xs + lrow * XP + lhi * 16;
      v4u alo = *(const v4u*)(ap2);
      v4u ahi = *(const v4u*)(ap2 + 8);
      const v16bf afr = make_frag(alo, ahi);
#pragma unroll
      for (int nt2 = 0; nt2 < 8; ++nt2) {
        const unsigned short* bp = W2p + (((nt2 * 16 + c2) * 32 + lane) << 4);
        v4u blo = *(const v4u*)(bp);
        v4u bhi = *(const v4u*)(bp + 8);
        acc2[nt2] = __builtin_amdgcn_wmma_f32_16x16x32_bf16(
            false, afr, false, make_frag(blo, bhi),
            (short)0, acc2[nt2], false, false);
      }
    }
  }

  // ---- bias + relu, stage X2 [16,128] bf16, then layer3 via WMMA ----
#pragma unroll
  for (int nt2 = 0; nt2 < 8; ++nt2) {
    const float bias = b2[nt2 * 16 + lrow];
#pragma unroll
    for (int r = 0; r < 8; ++r) {
      float v = fmaxf(acc2[nt2][r] + bias, 0.0f);
      xs[(r + lhi * 8) * XP + nt2 * 16 + lrow] = f2bf(v);
    }
  }

  v8f acc3 = v8f{0.f, 0.f, 0.f, 0.f, 0.f, 0.f, 0.f, 0.f};
#pragma unroll
  for (int kb = 0; kb < 4; ++kb) {
    const unsigned short* ap = xs + lrow * XP + kb * 32 + lhi * 16;
    v4u alo = *(const v4u*)(ap);
    v4u ahi = *(const v4u*)(ap + 8);
    const unsigned short* bp = W3p + ((kb * 32 + lane) << 4);
    v4u blo = *(const v4u*)(bp);
    v4u bhi = *(const v4u*)(bp + 8);
    acc3 = __builtin_amdgcn_wmma_f32_16x16x32_bf16(
        false, make_frag(alo, ahi), false, make_frag(blo, bhi),
        (short)0, acc3, false, false);
  }

  if (lrow == 0) {                        // only column 0 of D is the output
    const float b3v = b3[0];
    const int ebase = blockIdx.x * 128 + wave * 16 + lhi * 8;
#pragma unroll
    for (int r = 0; r < 8; ++r) {
      const int e = ebase + r;
      if (e < E) out[e] = acc3[r] + b3v;
    }
  }
}

// ---------------------------------------------------------------------------
extern "C" void kernel_launch(void* const* d_in, const int* in_sizes, int n_in,
                              void* d_out, int out_size, void* d_ws, size_t ws_size,
                              hipStream_t stream) {
  (void)n_in; (void)out_size; (void)ws_size;
  const float* emb = (const float*)d_in[0];
  const int*   ei  = (const int*)  d_in[1];
  const float* W1  = (const float*)d_in[2];
  const float* b1  = (const float*)d_in[3];
  const float* W2  = (const float*)d_in[4];
  const float* b2  = (const float*)d_in[5];
  const float* W3  = (const float*)d_in[6];
  const float* b3  = (const float*)d_in[7];
  float* out = (float*)d_out;
  const int E = in_sizes[1] / 2;

  unsigned short* W1p = (unsigned short*)d_ws;       // 32*8 frags  = 256 KB
  unsigned short* W2p = W1p + 32 * 8 * 512;          // 8*16 frags  = 128 KB
  unsigned short* W3p = W2p + 8 * 16 * 512;          // 1*4  frags  =   4 KB

  pack_weights<<<(32 * 8 * 512 + 255) / 256, 256, 0, stream>>>(W1, W1p, 256, 512, 8, 32);
  pack_weights<<<(8 * 16 * 512 + 255) / 256, 256, 0, stream>>>(W2, W2p, 512, 128, 16, 8);
  pack_weights<<<(1 * 4  * 512 + 255) / 256, 256, 0, stream>>>(W3, W3p, 128, 1, 4, 1);

  const int tiles = (E + 127) / 128;
  edge_mlp_wmma<<<tiles, 256, 0, stream>>>(emb, ei, W1p, W2p, W3p,
                                           b1, b2, b3, out, E);
}
